// GraphBlock_60533269070359
// MI455X (gfx1250) — matrix operand order repaired and checked by reading, not compiled, not run on hardware
//
#include <hip/hip_runtime.h>
#include <hip/hip_bf16.h>

// MI455X / gfx1250 implementation.
// All dense GEMMs go through v_wmma_f32_16x16x32_f16 (f16 in, f32 accum).
// Pipeline (all on `stream`):
//   k_cast x5  : f32 weights -> f16 scratch
//   k_ln1      : LN1(Z) -> Hh (f16)
//   k_qmix     : X = Hh@Wlin^T (f32), U0 = Z + Hh@Wmix^T   [WMMA]
//   k_attn     : GATv2 scores + softmax + attn@X per (b,h) in LDS -> Yh (f16)
//   k_wout     : U = U0 + Yh@Wout^T, node gate, LN2 -> U (f32), Vh (f16) [WMMA, fused epilogue]
//   k_mlp1     : Th = silu(Vh@W1^T + b1) (f16)             [WMMA]
//   k_mlp2     : out = U + Th@W2^T + b2                    [WMMA]
//
// gemm_strip is software-pipelined: all NT B-fragments of a k-step are loaded
// as one clause before the WMMA burst, so the compiler can wait with
// decreasing loadcnt thresholds instead of draining the queue per-WMMA.

typedef __attribute__((ext_vector_type(16))) _Float16 v16h;
typedef __attribute__((ext_vector_type(8)))  _Float16 v8h;
typedef __attribute__((ext_vector_type(8)))  float    v8f;

#define BB   128
#define NN   64
#define DD   128
#define HHN  4
#define DHH  32
#define DFF  512
#define MROWS (BB*NN)   // 8192

__device__ __forceinline__ float sigmoidf_(float x){ return 1.f/(1.f+__expf(-x)); }
__device__ __forceinline__ float siluf_(float x){ return x*sigmoidf_(x); }

// ---- WMMA fragment loaders (contiguous k order, same for A and B) ----

// A: 16x32 f16 tile, row-major A[lda], rows m0..m0+15, cols k0..k0+31.
// lane<16: M=lane, K=k0..k0+15; lane>=16: M=lane-16, K=k0+16..k0+31.
__device__ __forceinline__ v16h load_a_frag(const _Float16* A, int lda, int m0, int k0) {
  int lane = threadIdx.x & 31;
  int row  = m0 + (lane & 15);
  int kk   = k0 + ((lane >> 4) << 4);          // +0 or +16
  const _Float16* p = A + (size_t)row * lda + kk;
  v8h lo = *(const v8h*)(p);
  v8h hi = *(const v8h*)(p + 8);
  v16h f;
#pragma unroll
  for (int i = 0; i < 8; ++i) { f[i] = lo[i]; f[i + 8] = hi[i]; }
  return f;
}

// B: 32x16 f16 tile where B[k][n] = W[n][k] (W row-major, row length K).
// lane%16 = n, lane<16 -> K=k0..k0+15, lane>=16 -> K=k0+16..k0+31 (contiguous in W).
__device__ __forceinline__ v16h load_b_frag(const _Float16* W, int K, int n0, int k0) {
  int lane = threadIdx.x & 31;
  int col  = n0 + (lane & 15);
  int kk   = k0 + ((lane >> 4) << 4);          // +0 or +16
  return *(const v16h*)(W + (size_t)col * K + kk);
}

// One wave computes a 16 x (16*NT) strip of A@W^T with K-dim reduction.
// B fragments for a whole k-step are preloaded (one clause), then the WMMA
// burst issues against progressively-retiring loads; next A frag is hoisted.
template<int NT, int K>
__device__ __forceinline__ void gemm_strip(const _Float16* A, int lda, int m0,
                                           const _Float16* W, int n0, v8f c[NT]) {
  v16h a = load_a_frag(A, lda, m0, 0);
#pragma unroll
  for (int k = 0; k < K; k += 32) {
    v16h b[NT];
#pragma unroll
    for (int t = 0; t < NT; ++t) b[t] = load_b_frag(W, K, n0 + 16 * t, k);
    v16h a_next;
    if (k + 32 < K) {
      a_next = load_a_frag(A, lda, m0, k + 32);
      if (k + 64 < K)  // hint A chunk after next into cache: global_prefetch_b8
        __builtin_prefetch(A + (size_t)(m0 + (threadIdx.x & 15)) * lda + k + 64, 0, 1);
    }
#pragma unroll
    for (int t = 0; t < NT; ++t)
      c[t] = __builtin_amdgcn_wmma_f32_16x16x32_f16(false, a, false, b[t],
                                                    (short)0, c[t], false, false);
    if (k + 32 < K) a = a_next;
  }
}

// D tile layout: vgpr r -> row r + 8*(lane>=16), col lane%16.
#define TILE_STORE_LOOP(body)                                   \
  {                                                             \
    int n_l = lane & 15;                                        \
    int mb  = ((lane >> 4) << 3);                               \
    _Pragma("unroll") for (int t = 0; t < 8; ++t)               \
      _Pragma("unroll") for (int r = 0; r < 8; ++r) { body }    \
  }

// ---------------- kernels ----------------

__global__ void k_cast(const float* __restrict__ s, _Float16* __restrict__ d, int n) {
  int i = blockIdx.x * 256 + threadIdx.x;
  if (i < n) d[i] = (_Float16)s[i];
}

__global__ void k_ln1(const float* __restrict__ Z, const float* __restrict__ g,
                      const float* __restrict__ b, _Float16* __restrict__ Hh) {
  int wave = threadIdx.x >> 5, lane = threadIdx.x & 31;
  int row = blockIdx.x * 8 + wave;
  const float* z = Z + (size_t)row * DD;
  float v[4]; float s = 0.f, sq = 0.f;
#pragma unroll
  for (int i = 0; i < 4; ++i) { v[i] = z[lane + 32 * i]; s += v[i]; sq += v[i] * v[i]; }
#pragma unroll
  for (int off = 16; off > 0; off >>= 1) {
    s  += __shfl_xor(s,  off, 32);
    sq += __shfl_xor(sq, off, 32);
  }
  float mu  = s * (1.f / DD);
  float var = sq * (1.f / DD) - mu * mu;
  float rs  = rsqrtf(var + 1e-5f);
  _Float16* o = Hh + (size_t)row * DD;
#pragma unroll
  for (int i = 0; i < 4; ++i) {
    int cdx = lane + 32 * i;
    o[cdx] = (_Float16)((v[i] - mu) * rs * g[cdx] + b[cdx]);
  }
}

// blockIdx.y==0: X = Hh @ Wlin^T ; blockIdx.y==1: U0 = Z + Hh @ Wmix^T
__global__ void k_qmix(const _Float16* __restrict__ Hh, const _Float16* __restrict__ Wlh,
                       const _Float16* __restrict__ Wmh, const float* __restrict__ Z,
                       float* __restrict__ X, float* __restrict__ U0) {
  int wave = threadIdx.x >> 5, lane = threadIdx.x & 31;
  int m0 = blockIdx.x * 128 + wave * 16;
  bool mix = (blockIdx.y == 1);
  const _Float16* W = mix ? Wmh : Wlh;
  v8f zero = {0.f,0.f,0.f,0.f,0.f,0.f,0.f,0.f};
  v8f c[8];
#pragma unroll
  for (int t = 0; t < 8; ++t) c[t] = zero;
  gemm_strip<8, DD>(Hh, DD, m0, W, 0, c);
  TILE_STORE_LOOP({
    size_t gi = (size_t)(m0 + mb + r) * DD + t * 16 + n_l;
    if (mix) U0[gi] = Z[gi] + c[t][r];
    else     X[gi]  = c[t][r];
  })
}

// One block per (b,h): scores + softmax + attn@X fully in LDS.
__global__ void k_attn(const float* __restrict__ X, const float* __restrict__ avec,
                       const float* __restrict__ A0, const int* __restrict__ mask,
                       _Float16* __restrict__ Yh) {
  __shared__ float Xt[NN][DHH];   // 8 KB
  __shared__ float es[NN][NN];    // 16 KB
  __shared__ float as[DHH];
  int tid = threadIdx.x;
  int b = blockIdx.x >> 2, hh = blockIdx.x & 3;
  for (int q = tid; q < NN * DHH; q += 256) {
    int i = q >> 5, dd = q & 31;
    Xt[i][dd] = X[(size_t)(b * NN + i) * DD + hh * DHH + dd];
  }
  if (tid < DHH) as[tid] = avec[hh * DHH + tid];
  __syncthreads();
  for (int q = tid; q < NN * NN; q += 256) {
    int i = q >> 6, j = q & 63;
    float acc = 0.f;
#pragma unroll
    for (int dd = 0; dd < DHH; ++dd) {
      float v = Xt[i][dd] + Xt[j][dd];
      v = (v >= 0.f) ? v : 0.2f * v;           // LeakyReLU(0.2)
      acc += v * as[dd];
    }
    acc += __logf(A0[i * NN + j] + 1e-8f);
    es[i][j] = mask[i * NN + j] ? acc : -1e30f;
  }
  __syncthreads();
  if (tid < NN) {                               // row softmax
    float m = -1e30f;
    for (int j = 0; j < NN; ++j) m = fmaxf(m, es[tid][j]);
    float ssum = 0.f;
    for (int j = 0; j < NN; ++j) { float p = __expf(es[tid][j] - m); es[tid][j] = p; ssum += p; }
    float inv = 1.f / ssum;
    for (int j = 0; j < NN; ++j) es[tid][j] *= inv;
  }
  __syncthreads();
  for (int q = tid; q < NN * DHH; q += 256) {   // Y = attn @ X
    int i = q >> 5, dd = q & 31;
    float acc = 0.f;
    for (int j = 0; j < NN; ++j) acc += es[i][j] * Xt[j][dd];
    Yh[(size_t)(b * NN + i) * DD + hh * DHH + dd] = (_Float16)acc;
  }
}

// U = U0 + Yh@Wout^T, then squeeze-excite gate + LN2, fused (64 rows/block, 4 waves).
__global__ void k_wout(const _Float16* __restrict__ Yh, const _Float16* __restrict__ Wouth,
                       const float* __restrict__ U0,
                       const float* __restrict__ fc1w, const float* __restrict__ fc1b,
                       const float* __restrict__ fc2w, const float* __restrict__ fc2b,
                       const float* __restrict__ ln2g, const float* __restrict__ ln2b,
                       float* __restrict__ U, _Float16* __restrict__ Vh) {
  __shared__ float Ut[NN][DD];                  // 32 KB
  __shared__ float gs[NN], mus[NN], rss[NN];
  int tid = threadIdx.x, wave = tid >> 5, lane = tid & 31;
  int m0b = blockIdx.x * 64;
  int m0  = m0b + wave * 16;
  v8f zero = {0.f,0.f,0.f,0.f,0.f,0.f,0.f,0.f};
  v8f c[8];
#pragma unroll
  for (int t = 0; t < 8; ++t) c[t] = zero;
  gemm_strip<8, DD>(Yh, DD, m0, Wouth, 0, c);
  {
    int n_l = lane & 15;
    int mlb = wave * 16 + ((lane >> 4) << 3);
#pragma unroll
    for (int t = 0; t < 8; ++t)
#pragma unroll
      for (int r = 0; r < 8; ++r) {
        int col = t * 16 + n_l;
        Ut[mlb + r][col] = U0[(size_t)(m0b + mlb + r) * DD + col] + c[t][r];
      }
  }
  __syncthreads();
  if (tid < NN) {                               // per-row gate + LN2 stats
    float s = 0.f, sq = 0.f;
    for (int cc = 0; cc < DD; ++cc) { float v = Ut[tid][cc]; s += v; sq += v * v; }
    float mean = s * (1.f / DD);
    float acc = 0.f;
    for (int cc = 0; cc < DD; ++cc) acc += siluf_(mean * fc1w[cc] + fc1b[cc]) * fc2w[cc];
    float g  = sigmoidf_(acc + fc2b[0]);
    float mu = g * s * (1.f / DD);
    float var = g * g * sq * (1.f / DD) - mu * mu;
    gs[tid] = g; mus[tid] = mu; rss[tid] = rsqrtf(var + 1e-5f);
  }
  __syncthreads();
  for (int q = tid; q < NN * DD; q += 128) {
    int r = q >> 7, cc = q & 127;
    float v = Ut[r][cc] * gs[r];
    size_t gi = (size_t)(m0b + r) * DD + cc;
    U[gi]  = v;
    Vh[gi] = (_Float16)((v - mus[r]) * rss[r] * ln2g[cc] + ln2b[cc]);
  }
}

__global__ void k_mlp1(const _Float16* __restrict__ Vh, const _Float16* __restrict__ W1h,
                       const float* __restrict__ b1, _Float16* __restrict__ Th) {
  int wave = threadIdx.x >> 5, lane = threadIdx.x & 31;
  int m0  = blockIdx.x * 128 + wave * 16;
  int n0b = blockIdx.y * 128;
  v8f zero = {0.f,0.f,0.f,0.f,0.f,0.f,0.f,0.f};
  v8f c[8];
#pragma unroll
  for (int t = 0; t < 8; ++t) c[t] = zero;
  gemm_strip<8, DD>(Vh, DD, m0, W1h, n0b, c);
  TILE_STORE_LOOP({
    int ncol = n0b + t * 16 + n_l;
    Th[(size_t)(m0 + mb + r) * DFF + ncol] = (_Float16)siluf_(c[t][r] + b1[ncol]);
  })
}

__global__ void k_mlp2(const _Float16* __restrict__ Th, const _Float16* __restrict__ W2h,
                       const float* __restrict__ b2, const float* __restrict__ U,
                       float* __restrict__ out) {
  int wave = threadIdx.x >> 5, lane = threadIdx.x & 31;
  int m0 = blockIdx.x * 128 + wave * 16;
  v8f zero = {0.f,0.f,0.f,0.f,0.f,0.f,0.f,0.f};
  v8f c[8];
#pragma unroll
  for (int t = 0; t < 8; ++t) c[t] = zero;
  gemm_strip<8, DFF>(Th, DFF, m0, W2h, 0, c);
  TILE_STORE_LOOP({
    size_t gi = (size_t)(m0 + mb + r) * DD + t * 16 + n_l;
    out[gi] = U[gi] + c[t][r] + b2[t * 16 + n_l];
  })
}

// ---------------- host launcher ----------------

extern "C" void kernel_launch(void* const* d_in, const int* in_sizes, int n_in,
                              void* d_out, int out_size, void* d_ws, size_t ws_size,
                              hipStream_t stream) {
  (void)in_sizes; (void)n_in; (void)out_size; (void)ws_size;
  const float* Z    = (const float*)d_in[0];
  const int*   mask = (const int*)  d_in[1];
  const float* A0   = (const float*)d_in[2];
  const float* ln1g = (const float*)d_in[3];
  const float* ln1b = (const float*)d_in[4];
  const float* Wlin = (const float*)d_in[5];
  const float* avec = (const float*)d_in[6];
  const float* Wout = (const float*)d_in[7];
  const float* Wmix = (const float*)d_in[8];
  const float* fc1w = (const float*)d_in[9];
  const float* fc1b = (const float*)d_in[10];
  const float* fc2w = (const float*)d_in[11];
  const float* fc2b = (const float*)d_in[12];
  const float* ln2g = (const float*)d_in[13];
  const float* ln2b = (const float*)d_in[14];
  const float* W1   = (const float*)d_in[15];
  const float* b1   = (const float*)d_in[16];
  const float* W2   = (const float*)d_in[17];
  const float* b2   = (const float*)d_in[18];
  float* out = (float*)d_out;

  char* wsb = (char*)d_ws;
  size_t off = 0;
  auto alloc = [&](size_t bytes) -> void* {
    void* p = wsb + off;
    off = (off + bytes + 255) & ~(size_t)255;
    return p;
  };
  _Float16* Hh    = (_Float16*)alloc((size_t)MROWS * DD  * 2);
  _Float16* Wlh   = (_Float16*)alloc((size_t)DD * DD     * 2);
  _Float16* Wmh   = (_Float16*)alloc((size_t)DD * DD     * 2);
  _Float16* Wouth = (_Float16*)alloc((size_t)DD * DD     * 2);
  _Float16* W1h   = (_Float16*)alloc((size_t)DFF * DD    * 2);
  _Float16* W2h   = (_Float16*)alloc((size_t)DD * DFF    * 2);
  float*    Xb    = (float*)   alloc((size_t)MROWS * DD  * 4);
  float*    U0    = (float*)   alloc((size_t)MROWS * DD  * 4);
  _Float16* Yh    = (_Float16*)alloc((size_t)MROWS * DD  * 2);
  float*    U     = (float*)   alloc((size_t)MROWS * DD  * 4);
  _Float16* Vh    = (_Float16*)alloc((size_t)MROWS * DD  * 2);
  _Float16* Th    = (_Float16*)alloc((size_t)MROWS * DFF * 2);

  k_cast<<<dim3((DD * DD  + 255) / 256), 256, 0, stream>>>(Wlin, Wlh, DD * DD);
  k_cast<<<dim3((DD * DD  + 255) / 256), 256, 0, stream>>>(Wmix, Wmh, DD * DD);
  k_cast<<<dim3((DD * DD  + 255) / 256), 256, 0, stream>>>(Wout, Wouth, DD * DD);
  k_cast<<<dim3((DFF * DD + 255) / 256), 256, 0, stream>>>(W1, W1h, DFF * DD);
  k_cast<<<dim3((DD * DFF + 255) / 256), 256, 0, stream>>>(W2, W2h, DD * DFF);

  k_ln1 <<<MROWS / 8, 256, 0, stream>>>(Z, ln1g, ln1b, Hh);
  k_qmix<<<dim3(MROWS / 128, 2), 256, 0, stream>>>(Hh, Wlh, Wmh, Z, Xb, U0);
  k_attn<<<BB * HHN, 256, 0, stream>>>(Xb, avec, A0, mask, Yh);
  k_wout<<<MROWS / 64, 128, 0, stream>>>(Yh, Wouth, U0, fc1w, fc1b, fc2w, fc2b,
                                         ln2g, ln2b, U, Vh);
  k_mlp1<<<dim3(MROWS / 128, DFF / 128), 256, 0, stream>>>(Vh, W1h, b1, Th);
  k_mlp2<<<MROWS / 128, 256, 0, stream>>>(Th, W2h, b2, U, out);
}
